// BahdanauAttention_60404420051398
// MI455X (gfx1250) — matrix operand order repaired and checked by reading, not compiled, not run on hardware
//
#include <hip/hip_runtime.h>
#include <hip/hip_bf16.h>
#include <cstdint>
#include <cstddef>

// Problem constants (match reference): B=32, S=2048, K=2*EH=1024, A=512
#define NB 32
#define NS 2048
#define NK 1024
#define NA 512

typedef __attribute__((ext_vector_type(16))) __bf16 v16bf;
typedef __attribute__((ext_vector_type(8)))  float v8f;

// round-to-nearest-even fp32 -> bf16 (bit pattern)
__device__ __forceinline__ unsigned short f2bf(float f) {
    unsigned int u = __float_as_uint(f);
    u += 0x7FFFu + ((u >> 16) & 1u);
    return (unsigned short)(u >> 16);
}

// branch-free tanh: prefer CDNA5 V_TANH trans op when the builtin exists
__device__ __forceinline__ float fast_tanh(float x) {
#if __has_builtin(__builtin_amdgcn_tanhf)
    return __builtin_amdgcn_tanhf(x);
#else
    const float e = __expf(2.0f * x);
    return 1.0f - 2.0f / (e + 1.0f);
#endif
}

// ---------------------------------------------------------------------------
// Kernel 1: convert W (1024x512 fp32, row-major [K][N]) into bf16 fragment-
// major layout. One WMMA B-fragment = 16x32 tile (N-tile nt, K-step ks).
// Per fragment: 32 lanes x 8 dwords; lane l holds column n = nt*16 + (l&15),
// K half = (l>>4)*16, dword i packs bf16(K=base+2i), bf16(K=base+2i+1).
// Fragment f = nt*32+ks occupies 1KB; lane data is 32 contiguous bytes.
// ---------------------------------------------------------------------------
__global__ __launch_bounds__(256) void prep_w_kernel(const float* __restrict__ W,
                                                     uint32_t* __restrict__ Wp) {
    const int blk = blockIdx.x;          // fragment id: nt*32 + ks (1024 total)
    const int tid = threadIdx.x;         // lane*8 + i
    const int nt = blk >> 5, ks = blk & 31;
    const int lane = tid >> 3, i = tid & 7;
    const int n = nt * 16 + (lane & 15);
    const int k = ks * 32 + (lane >> 4) * 16 + 2 * i;
    const float w0 = W[(size_t)k * NA + n];
    const float w1 = W[(size_t)(k + 1) * NA + n];
    Wp[(size_t)blk * 256 + tid] = (uint32_t)f2bf(w0) | ((uint32_t)f2bf(w1) << 16);
}

// ---------------------------------------------------------------------------
// Kernel 2: temp2 = dh @ U, where dh[b][k] = decoder_hidden[k<512?0:1][b][k%512]
// (32 x 512 outputs, 1024-length dots) — tiny, plain fp32.
// ---------------------------------------------------------------------------
__global__ __launch_bounds__(256) void temp2_kernel(const float* __restrict__ dec,
                                                    const float* __restrict__ U,
                                                    float* __restrict__ temp2) {
    const int idx = blockIdx.x * 256 + threadIdx.x;   // 64 blocks -> 16384
    const int b = idx >> 9;
    const int a = idx & 511;
    const float* d0 = dec + (size_t)b * 512;                     // layer 0 slice
    const float* d1 = dec + (size_t)NB * 512 + (size_t)b * 512;  // layer 1 slice
    float acc = 0.f;
    for (int k = 0; k < 512; ++k) acc = fmaf(d0[k], U[(size_t)k * NA + a], acc);
    for (int k = 0; k < 512; ++k) acc = fmaf(d1[k], U[(size_t)(512 + k) * NA + a], acc);
    temp2[idx] = acc;
}

// ---------------------------------------------------------------------------
// Kernel 3: fused score pass.
//   scores[b][s] = tanh(enc[b,s,:] @ W + temp2[b,:]) . v
// One 256-thread WG per (b, 32 rows of S): M-tile = 32 (two 16-row sub-tiles)
// so every B-fragment fetched from L2 feeds TWO wmma ops (B reuse = 2,
// halving L2 traffic vs a 16-row tile). A-tile staged to LDS as bf16 with a
// padded row stride of 1032 elements (conflict-free ds_load_b128). 8 waves
// split N=512 into 64-wide ranges; 256 v_wmma_f32_16x16x32_bf16 per wave.
// ---------------------------------------------------------------------------
__global__ __launch_bounds__(256) void score_kernel(const float* __restrict__ enc,
                                                    const uint32_t* __restrict__ Wp,
                                                    const float* __restrict__ temp2,
                                                    const float* __restrict__ vvec,
                                                    float* __restrict__ scores) {
    __shared__ unsigned short smA[32 * 1032];   // 66,048 B, bf16 A tile, padded rows
    __shared__ float smScore[32];

    const int tid = threadIdx.x;
    const int b  = blockIdx.x >> 6;           // 64 s-tiles of 32 per batch row
    const int s0 = (blockIdx.x & 63) << 5;

    // ---- stage A tile: 32x1024 fp32 -> bf16, fully-coalesced float4 stream
    const float* tile = enc + ((size_t)(b * NS + s0)) * NK;
    #pragma unroll
    for (int j = 0; j < 32; ++j) {
        const int off = (j * 256 + tid) * 4;          // flat float index in 32x1024
        const float4 f = *reinterpret_cast<const float4*>(tile + off);
        const int m = off >> 10, k = off & 1023;
        uint32_t* dst = reinterpret_cast<uint32_t*>(&smA[m * 1032 + k]);
        dst[0] = (uint32_t)f2bf(f.x) | ((uint32_t)f2bf(f.y) << 16);
        dst[1] = (uint32_t)f2bf(f.z) | ((uint32_t)f2bf(f.w) << 16);
    }
    if (tid < 32) smScore[tid] = 0.f;
    __syncthreads();

    const int wave = tid >> 5;
    const int lane = tid & 31;
    const int m16 = lane & 15;
    const int h   = lane >> 4;

    v8f acc[2][4] = {};   // [m sub-tile][n tile], 16x16 f32 accumulators

    const uint4* wp4 = reinterpret_cast<const uint4*>(Wp);
    for (int ks = 0; ks < 32; ++ks) {
        // A fragments: lane m=m16, K half h;
        // elements 0-15 at kbase+16h..kbase+16h+15 (matches B packing).
        const int kbase = ks * 32;
        union { uint4 q[2]; v16bf v; } A0, A1;
        {
            const uint4* ap0 = reinterpret_cast<const uint4*>(&smA[m16 * 1032 + kbase + 16 * h]);
            const uint4* ap1 = reinterpret_cast<const uint4*>(&smA[(16 + m16) * 1032 + kbase + 16 * h]);
            A0.q[0] = ap0[0]; A0.q[1] = ap0[1];   // +0B and +16B chunks
            A1.q[0] = ap1[0]; A1.q[1] = ap1[1];
        }
        #pragma unroll
        for (int t = 0; t < 4; ++t) {
            const int nt = wave * 4 + t;
            const uint4* bp = wp4 + ((size_t)(nt * 32 + ks) * 64 + lane * 2);
            union { uint4 q[2]; v16bf v; } Bm;
            Bm.q[0] = bp[0];
            Bm.q[1] = bp[1];
            acc[0][t] = __builtin_amdgcn_wmma_f32_16x16x32_bf16(
                false, A0.v, false, Bm.v, (short)0, acc[0][t], false, false);
            acc[1][t] = __builtin_amdgcn_wmma_f32_16x16x32_bf16(
                false, A1.v, false, Bm.v, (short)0, acc[1][t], false, false);
        }
    }

    // ---- epilogue: tanh(acc + temp2[n]) * v[n], reduce over n per row m
    // C layout: VGPR r, lanes 0-15 -> (M=r, N=lane); lanes 16-31 -> (M=8+r).
    float part[2][8];
    #pragma unroll
    for (int mt = 0; mt < 2; ++mt)
        #pragma unroll
        for (int r = 0; r < 8; ++r) part[mt][r] = 0.f;

    #pragma unroll
    for (int t = 0; t < 4; ++t) {
        const int n = (wave * 4 + t) * 16 + m16;
        const float t2 = temp2[b * NA + n];
        const float vn = vvec[n];
        #pragma unroll
        for (int r = 0; r < 8; ++r) {
            part[0][r] += fast_tanh(acc[0][t][r] + t2) * vn;
            part[1][r] += fast_tanh(acc[1][t][r] + t2) * vn;
        }
    }
    #pragma unroll
    for (int mt = 0; mt < 2; ++mt) {
        #pragma unroll
        for (int r = 0; r < 8; ++r) {
            float s = part[mt][r];
            s += __shfl_xor(s, 1, 16);
            s += __shfl_xor(s, 2, 16);
            s += __shfl_xor(s, 4, 16);
            s += __shfl_xor(s, 8, 16);
            if (m16 == 0) atomicAdd(&smScore[mt * 16 + h * 8 + r], s);  // ds_add_f32
        }
    }
    __syncthreads();
    if (tid < 32) scores[(size_t)b * NS + s0 + tid] = smScore[tid];
}

// ---------------------------------------------------------------------------
// Kernel 4: softmax over S per batch row. 1 WG per row.
// ---------------------------------------------------------------------------
__global__ __launch_bounds__(256) void softmax_kernel(const float* __restrict__ scores,
                                                      float* __restrict__ attn) {
    __shared__ float red[256];
    const int b = blockIdx.x, tid = threadIdx.x;
    const float* row = scores + (size_t)b * NS;
    float v[8];
    float mx = -3.402823466e38f;
    #pragma unroll
    for (int j = 0; j < 8; ++j) { v[j] = row[tid + 256 * j]; mx = fmaxf(mx, v[j]); }
    red[tid] = mx; __syncthreads();
    for (int s = 128; s > 0; s >>= 1) {
        if (tid < s) red[tid] = fmaxf(red[tid], red[tid + s]);
        __syncthreads();
    }
    mx = red[0]; __syncthreads();
    float sum = 0.f;
    #pragma unroll
    for (int j = 0; j < 8; ++j) { v[j] = __expf(v[j] - mx); sum += v[j]; }
    red[tid] = sum; __syncthreads();
    for (int s = 128; s > 0; s >>= 1) {
        if (tid < s) red[tid] += red[tid + s];
        __syncthreads();
    }
    const float inv = 1.f / red[0];
    #pragma unroll
    for (int j = 0; j < 8; ++j) attn[(size_t)b * NS + tid + 256 * j] = v[j] * inv;
}

// ---------------------------------------------------------------------------
// Kernel 5: context[b][e] = sum_s attn[b][s] * enc[b][s][e].
// 4 WGs per batch row (256 e-columns each); weights staged in LDS; enc streamed
// fully coalesced (1KB per step). HBM-bound by design.
// ---------------------------------------------------------------------------
__global__ __launch_bounds__(256) void context_kernel(const float* __restrict__ attn,
                                                      const float* __restrict__ enc,
                                                      float* __restrict__ ctx) {
    __shared__ float wsh[NS];
    const int b = blockIdx.x >> 2;
    const int e = ((blockIdx.x & 3) << 8) + threadIdx.x;
    #pragma unroll
    for (int j = 0; j < 8; ++j)
        wsh[threadIdx.x + 256 * j] = attn[(size_t)b * NS + threadIdx.x + 256 * j];
    __syncthreads();
    const float* base = enc + (size_t)b * NS * NK + e;
    float a0 = 0.f, a1 = 0.f, a2 = 0.f, a3 = 0.f;
    for (int s = 0; s < NS; s += 4) {
        a0 = fmaf(wsh[s + 0], base[(size_t)(s + 0) * NK], a0);
        a1 = fmaf(wsh[s + 1], base[(size_t)(s + 1) * NK], a1);
        a2 = fmaf(wsh[s + 2], base[(size_t)(s + 2) * NK], a2);
        a3 = fmaf(wsh[s + 3], base[(size_t)(s + 3) * NK], a3);
    }
    ctx[(size_t)b * NK + e] = (a0 + a1) + (a2 + a3);
}

// ---------------------------------------------------------------------------
extern "C" void kernel_launch(void* const* d_in, const int* in_sizes, int n_in,
                              void* d_out, int out_size, void* d_ws, size_t ws_size,
                              hipStream_t stream) {
    (void)in_sizes; (void)n_in; (void)out_size; (void)ws_size;
    const float* dec = (const float*)d_in[0];   // (2, 32, 512)
    const float* enc = (const float*)d_in[1];   // (32, 2048, 1024)
    const float* W   = (const float*)d_in[2];   // (1024, 512)
    const float* U   = (const float*)d_in[3];   // (1024, 512)
    const float* v   = (const float*)d_in[4];   // (512, 1)

    float* out_ctx  = (float*)d_out;            // context: 32*1024 floats
    float* out_attn = out_ctx + NB * NK;        // attention weights: 32*2048

    uint8_t* ws = (uint8_t*)d_ws;
    uint32_t* Wp  = (uint32_t*)ws;                               // 1 MB bf16 fragments
    float* temp2  = (float*)(ws + (1u << 20));                   // 64 KB
    float* scores = (float*)(ws + (1u << 20) + (1u << 16));      // 256 KB

    hipLaunchKernelGGL(prep_w_kernel,   dim3(1024), dim3(256), 0, stream, W, Wp);
    hipLaunchKernelGGL(temp2_kernel,    dim3(64),   dim3(256), 0, stream, dec, U, temp2);
    hipLaunchKernelGGL(score_kernel,    dim3(2048), dim3(256), 0, stream, enc, Wp, temp2, v, scores);
    hipLaunchKernelGGL(softmax_kernel,  dim3(NB),   dim3(256), 0, stream, scores, out_attn);
    hipLaunchKernelGGL(context_kernel,  dim3(NB*4), dim3(256), 0, stream, out_attn, enc, out_ctx);
}
